// ManualRNN_59528246722696
// MI455X (gfx1250) — matrix operand order repaired and checked by reading, not compiled, not run on hardware
//
#include <hip/hip_runtime.h>
#include <hip/hip_bf16.h>
#include <math.h>

#define B_ 64
#define S_ 512
#define I_ 256
#define H_ 1024
#define O_ 256
#define RWG 64   // workgroups in persistent recurrence kernel

typedef __attribute__((ext_vector_type(16))) _Float16 v16h;
typedef __attribute__((ext_vector_type(8)))  _Float16 v8h;
typedef __attribute__((ext_vector_type(8)))  float    v8f;

// ---------------- fragment loaders (contiguous k = 16*half + s, same order for A and B) ----------------
// A 16x32 f16 (MxK): lane lo(0-15): M=lane, K = [0..15]
//                    lane hi(16-31): M=lane-16, K = [16..31]
__device__ __forceinline__ v16h load_a_frag(const _Float16* __restrict__ A,
                                            int lda, int m0, int k0, int lane) {
  const int lo = lane & 15, hi = lane >> 4;
  const _Float16* p = A + (size_t)(m0 + lo) * lda + k0 + hi * 16;
  union { v16h v; v8h h[2]; } u;
  u.h[0] = *(const v8h*)(p);
  u.h[1] = *(const v8h*)(p + 8);
  return u.v;
}

// B 32x16 f16 (KxN), B[k][n] = W[n][k] with W row-major [N,K]:
// lane lo: N=lane, K=k0..k0+15 ; lane hi: N=lane-16, K=k0+16..k0+31
__device__ __forceinline__ v16h load_b_frag_g(const _Float16* __restrict__ W,
                                              int ldw, int n0, int k0, int lane) {
  const int lo = lane & 15, hi = lane >> 4;
  return *(const v16h*)(W + (size_t)(n0 + lo) * ldw + k0 + hi * 16);
}

__device__ __forceinline__ v16h load_b_frag_lds(const _Float16* Wl,
                                                int ldw, int k0, int lane) {
  const int lo = lane & 15, hi = lane >> 4;
  return *(const v16h*)(Wl + lo * ldw + k0 + hi * 16);
}

#define WMMA_F32_F16(a, b, c) \
  __builtin_amdgcn_wmma_f32_16x16x32_f16(false, (a), false, (b), (short)0, (c), false, false)

// ---------------- init: zero hs[0], h f16 buffer 0, barrier counter ----------------
__global__ __launch_bounds__(256) void init_kernel(float* hs0, _Float16* h0buf, unsigned* counter) {
  int i = blockIdx.x * blockDim.x + threadIdx.x;
  if (i < B_ * H_) { hs0[i] = 0.0f; h0buf[i] = (_Float16)0.0f; }
  if (i == 0) *counter = 0u;
}

// ---------------- convert x [B,S,I] fp32 -> xf16 [(s*B+b), I] f16 (permuted) ----------------
__global__ __launch_bounds__(256) void convert_x_kernel(const float* __restrict__ x,
                                                        _Float16* __restrict__ xf) {
  int o = blockIdx.x * blockDim.x + threadIdx.x;   // 0 .. 32768*256-1
  int i = o & (I_ - 1);
  int r = o >> 8;             // r = s*64 + b
  int s = r >> 6;
  int b = r & 63;
  xf[o] = (_Float16)x[((size_t)b * S_ + s) * I_ + i];
}

// ---------------- generic fp32 -> f16 ----------------
__global__ __launch_bounds__(256) void convert_f16_kernel(const float* __restrict__ src,
                                                          _Float16* __restrict__ dst, int n) {
  int i = blockIdx.x * blockDim.x + threadIdx.x;
  if (i < n) dst[i] = (_Float16)src[i];
}

// ---------------- x_proj GEMM: [32768,256] x [256,1024] -> hidden slots 1..512 ----------------
// Each wave computes a 16x64 strip (4 N-tiles sharing one A fragment).
// out row r = s*64+b goes to hid[(r + 64)*1024 + n]  (== slot s+1, batch b)
__global__ __launch_bounds__(256) void xproj_kernel(const _Float16* __restrict__ xf,
                                                    const _Float16* __restrict__ wih,
                                                    float* __restrict__ hid) {
  const int lane = threadIdx.x & 31;
  const int wave = threadIdx.x >> 5;
  const int tile = blockIdx.x * 8 + wave;      // 2048 M-tiles x 16 N-groups = 32768
  const int mt = tile >> 4;                    // M tile
  const int ng = tile & 15;                    // N group of 64
  const int m0 = mt * 16, n0 = ng * 64;
  v8f acc[4] = {{}, {}, {}, {}};
#pragma unroll
  for (int k0 = 0; k0 < I_; k0 += 32) {
    v16h a = load_a_frag(xf, I_, m0, k0, lane);
    v16h bfr[4];
#pragma unroll
    for (int j = 0; j < 4; ++j)
      bfr[j] = load_b_frag_g(wih, I_, n0 + j * 16, k0, lane);
#pragma unroll
    for (int j = 0; j < 4; ++j)
      acc[j] = WMMA_F32_F16(a, bfr[j], acc[j]);
  }
  const int lo = lane & 15, hi = lane >> 4;
#pragma unroll
  for (int j = 0; j < 4; ++j) {
    const int n = n0 + j * 16 + lo;
#pragma unroll
    for (int v = 0; v < 8; ++v) {
      int r = m0 + v + hi * 8;
      hid[(size_t)(r + B_) * H_ + n] = acc[j][v];
    }
  }
}

// ---------------- persistent recurrence kernel ----------------
// 64 WGs x 128 threads. WG g owns output columns n0 = g*16 .. +15 and keeps its
// W_hh slice (16 rows x 1024, f16, 32KB) resident in LDS for all 512 steps.
// Two independent WMMA accumulation chains halve the serial C->C depth per step.
__global__ __launch_bounds__(128) void rnn_kernel(const _Float16* __restrict__ whh,
                                                  const float* __restrict__ bias,
                                                  float* __restrict__ hid,
                                                  _Float16* __restrict__ h0buf,
                                                  _Float16* __restrict__ h1buf,
                                                  unsigned* __restrict__ counter) {
  __shared__ _Float16 wsl[16 * H_];            // 32 KB
  const int tid  = threadIdx.x;
  const int lane = tid & 31;
  const int wave = tid >> 5;
  const int n0   = blockIdx.x * 16;

  // preload W_hh rows n0..n0+15 (contiguous 32KB region) into LDS
  {
    const uint4* src = (const uint4*)(whh + (size_t)n0 * H_);
    uint4* dst = (uint4*)wsl;
    for (int i = tid; i < (16 * H_) / 8; i += 128) dst[i] = src[i];
  }
  __syncthreads();

  const int m0 = wave * 16;                    // batch tile per wave
  const int lo = lane & 15, hi = lane >> 4;
  const int n  = n0 + lo;
  const float bn = bias[n];

  // per-lane element offsets into a [B,H] slab (8 C-tile rows this lane owns)
  size_t offs[8];
#pragma unroll
  for (int v = 0; v < 8; ++v) offs[v] = (size_t)(m0 + v + hi * 8) * H_ + n;

  for (int t = 0; t < S_; ++t) {
    const _Float16* hcur = (t & 1) ? h1buf : h0buf;
    _Float16*       hnxt = (t & 1) ? h0buf : h1buf;
    float* xp = hid + (size_t)(t + 1) * (B_ * H_);

    // x_proj values for this step are independent of the GEMM: issue loads first
    float xpv[8];
#pragma unroll
    for (int v = 0; v < 8; ++v) xpv[v] = xp[offs[v]];

    // prefetch next step's x_proj lines while this step's GEMM runs
    if (t + 1 < S_) {
      const float* xpn = xp + (B_ * H_);
#pragma unroll
      for (int v = 0; v < 8; ++v) __builtin_prefetch(&xpn[offs[v]], 0, 1);
    }

    // dual accumulation chains over K (even/odd 32-chunks)
    v8f acc0 = {};
    v8f acc1 = {};
#pragma unroll 4
    for (int k0 = 0; k0 < H_; k0 += 64) {
      v16h a0 = load_a_frag(hcur, H_, m0, k0, lane);
      v16h b0 = load_b_frag_lds(wsl, H_, k0, lane);
      v16h a1 = load_a_frag(hcur, H_, m0, k0 + 32, lane);
      v16h b1 = load_b_frag_lds(wsl, H_, k0 + 32, lane);
      acc0 = WMMA_F32_F16(a0, b0, acc0);
      acc1 = WMMA_F32_F16(a1, b1, acc1);
    }

    // h_new = tanh(acc + x_proj + bias); overwrite fp32 slot in place, stash f16 copy
#pragma unroll
    for (int v = 0; v < 8; ++v) {
      float val = tanhf(acc0[v] + acc1[v] + xpv[v] + bn);
      xp[offs[v]] = val;
      hnxt[offs[v]] = (_Float16)val;
    }

    // grid-wide barrier (monotonic counter, agent scope, release/acquire)
    __syncthreads();
    if (tid == 0) {
      unsigned target = (unsigned)(t + 1) * RWG;
      __hip_atomic_fetch_add(counter, 1u, __ATOMIC_ACQ_REL, __HIP_MEMORY_SCOPE_AGENT);
      while (__hip_atomic_load(counter, __ATOMIC_ACQUIRE, __HIP_MEMORY_SCOPE_AGENT) < target)
        __builtin_amdgcn_s_sleep(1);
    }
    __syncthreads();
  }
}

// ---------------- final FC: [64,1024] x [1024,256] + bias ----------------
__global__ __launch_bounds__(256) void fc_kernel(const _Float16* __restrict__ hlast,
                                                 const _Float16* __restrict__ fcw,
                                                 const float* __restrict__ fcb,
                                                 float* __restrict__ out) {
  const int lane = threadIdx.x & 31;
  const int wave = threadIdx.x >> 5;
  const int tile = blockIdx.x * 8 + wave;      // 64 tiles: 4 M x 16 N
  const int mt = tile >> 4;
  const int nt = tile & 15;
  const int m0 = mt * 16, n0 = nt * 16;
  v8f acc0 = {};
  v8f acc1 = {};
#pragma unroll 4
  for (int k0 = 0; k0 < H_; k0 += 64) {
    v16h a0 = load_a_frag(hlast, H_, m0, k0, lane);
    v16h b0 = load_b_frag_g(fcw, H_, n0, k0, lane);
    v16h a1 = load_a_frag(hlast, H_, m0, k0 + 32, lane);
    v16h b1 = load_b_frag_g(fcw, H_, n0, k0 + 32, lane);
    acc0 = WMMA_F32_F16(a0, b0, acc0);
    acc1 = WMMA_F32_F16(a1, b1, acc1);
  }
  const int lo = lane & 15, hi = lane >> 4;
  const int n = n0 + lo;
  const float bn = fcb[n];
#pragma unroll
  for (int v = 0; v < 8; ++v) {
    int b = m0 + v + hi * 8;
    out[(size_t)b * O_ + n] = acc0[v] + acc1[v] + bn;
  }
}

// ---------------- workspace layout (bytes) ----------------
#define XF_OFF   0u                     // 32768*256 f16 = 16 MB
#define WIH_OFF  16777216u              // 1024*256 f16  = 512 KB
#define WHH_OFF  17301504u              // 1024*1024 f16 = 2 MB
#define FCW_OFF  19398656u              // 256*1024 f16  = 512 KB
#define H0_OFF   19922944u              // 64*1024 f16   = 128 KB
#define H1_OFF   20054016u              // 64*1024 f16   = 128 KB
#define CNT_OFF  20185088u              // 4 bytes

extern "C" void kernel_launch(void* const* d_in, const int* in_sizes, int n_in,
                              void* d_out, int out_size, void* d_ws, size_t ws_size,
                              hipStream_t stream) {
  const float* x    = (const float*)d_in[0];
  const float* wih  = (const float*)d_in[1];
  const float* whh  = (const float*)d_in[2];
  const float* bias = (const float*)d_in[3];
  const float* fcw  = (const float*)d_in[4];
  const float* fcb  = (const float*)d_in[5];

  float* out = (float*)d_out;          // [64,256] then hidden_states [513,64,1024]
  float* hid = out + B_ * O_;

  char* ws = (char*)d_ws;
  _Float16* xf   = (_Float16*)(ws + XF_OFF);
  _Float16* wihf = (_Float16*)(ws + WIH_OFF);
  _Float16* whhf = (_Float16*)(ws + WHH_OFF);
  _Float16* fcwf = (_Float16*)(ws + FCW_OFF);
  _Float16* h0b  = (_Float16*)(ws + H0_OFF);
  _Float16* h1b  = (_Float16*)(ws + H1_OFF);
  unsigned* cnt  = (unsigned*)(ws + CNT_OFF);

  init_kernel<<<(B_ * H_ + 255) / 256, 256, 0, stream>>>(hid, h0b, cnt);
  convert_x_kernel<<<(S_ * B_ * I_) / 256, 256, 0, stream>>>(x, xf);
  convert_f16_kernel<<<(H_ * I_ + 255) / 256, 256, 0, stream>>>(wih, wihf, H_ * I_);
  convert_f16_kernel<<<(H_ * H_ + 255) / 256, 256, 0, stream>>>(whh, whhf, H_ * H_);
  convert_f16_kernel<<<(O_ * H_ + 255) / 256, 256, 0, stream>>>(fcw, fcwf, O_ * H_);

  // x_proj: 2048 M-tiles x 16 N-groups, 8 waves/WG -> 4096 WGs
  xproj_kernel<<<4096, 256, 0, stream>>>(xf, wihf, hid);

  // recurrence: 64 persistent WGs, 512 steps with grid barrier.
  // Final h (f16) lands in buffer 0 (512 is even).
  rnn_kernel<<<RWG, 128, 0, stream>>>(whhf, bias, hid, h0b, h1b, cnt);

  // output head
  fc_kernel<<<8, 256, 0, stream>>>(h0b, fcwf, fcb, out);
}
